// TreeTransformerCell_topdown_86363202388293
// MI455X (gfx1250) — matrix operand order repaired and checked by reading, not compiled, not run on hardware
//
#include <hip/hip_runtime.h>
#include <hip/hip_bf16.h>
#include <cstdint>
#include <math.h>

#define D_MODEL 256
#define D_FF    512
#define TILE_M  32
#define LN_EPS  1e-5f

// LDS layout (dynamic, 80 KB total)
#define CTX_F_OFF  0                    // [32][256] f32   = 32768 B
#define CTX_BT_OFF 32768                // 16 A-frag tiles = 16384 B (2 mt x 8 ko x 1KB)
#define HID_BT_OFF 49152                // 32 A-frag tiles = 32768 B (2 mt x 16 ko x 1KB)
#define LDS_BYTES  81920

typedef __attribute__((ext_vector_type(16))) __bf16 v16bf;
typedef __attribute__((ext_vector_type(8)))  float  v8f;

union FragAB { v16bf v; uint32_t u[8]; uint4 q[2]; };

__device__ __forceinline__ uint16_t f2bf(float f) {
    uint32_t u = __float_as_uint(f);
    u += 0x7FFFu + ((u >> 16) & 1u);   // round-to-nearest-even
    return (uint16_t)(u >> 16);
}
__device__ __forceinline__ uint32_t pk2bf(float a, float b) {
    return (uint32_t)f2bf(a) | ((uint32_t)f2bf(b) << 16);
}

__device__ __forceinline__ float wsum(float v) {
#pragma unroll
    for (int m = 16; m >= 1; m >>= 1) v += __shfl_xor(v, m, 32);
    return v;
}

// Exact-form GELU: 0.5*x*(1+erf(x/sqrt(2))).
// erf via branch-free Abramowitz-Stegun 7.1.26 (|err| < 1.5e-7): uses the
// native v_rcp_f32 / v_exp_f32 transcendentals, no exec-mask divergence.
__device__ __forceinline__ float gelu_exact(float x) {
    float z = x * 0.70710678118f;
    float a = fabsf(z);
    float t = __builtin_amdgcn_rcpf(fmaf(0.3275911f, a, 1.0f));
    float p = fmaf(1.061405429f, t, -1.453152027f);
    p = fmaf(p, t, 1.421413741f);
    p = fmaf(p, t, -0.284496736f);
    p = fmaf(p, t, 0.254829592f);
    p = p * t;
    float e = __expf(-z * z);                 // v_exp_f32 (underflows to 0: fine)
    float erf_z = copysignf(fmaf(-p, e, 1.0f), z);
    return 0.5f * x * (1.0f + erf_z);
}

// ---------------------------------------------------------------------------
// out = h (rows not overwritten by the scatter keep their value)
// ---------------------------------------------------------------------------
__global__ void copy_h(const float4* __restrict__ src, float4* __restrict__ dst,
                       long n4) {
    long i = (long)blockIdx.x * blockDim.x + threadIdx.x;
    long stride = (long)gridDim.x * blockDim.x;
    for (; i < n4; i += stride) dst[i] = src[i];
}

// ---------------------------------------------------------------------------
// Pack W1 [256x512] / W2 [512x256] (f32 row-major, K-major rows) into bf16
// WMMA B-fragment tiles. Tile(ko,nt): K in [ko*32,+32), N in [nt*16,+16).
// Lane: col n = nt*16 + (lane&15); K base = ko*32 + (lane>>4)*16; dword i =
// bf16(K=base+2i) | bf16(K=base+2i+1)<<16.  Tile = 32 lanes x 32 B = 1 KB.
// W1: tiles 0..255 at ws[0]; W2: tiles 256..511 (ws + 256 KB).
// ---------------------------------------------------------------------------
__global__ void pack_weights(const float* __restrict__ W1,
                             const float* __restrict__ W2,
                             uint32_t* __restrict__ ws) {
    int t    = blockIdx.x * blockDim.x + threadIdx.x;
    int lane = t & 31;
    int tile = t >> 5;
    if (tile < 256) {                       // W1: ko in [0,8), nt in [0,32)
        int ko = tile >> 5, nt = tile & 31;
        int n  = nt * 16 + (lane & 15);
        int kb = ko * 32 + ((lane >> 4) << 4);
        uint32_t* dst = ws + tile * 256 + lane * 8;
#pragma unroll
        for (int i = 0; i < 8; ++i) {
            int k = kb + 2 * i;
            dst[i] = pk2bf(W1[(long)k * D_FF + n], W1[(long)(k + 1) * D_FF + n]);
        }
    } else if (tile < 512) {                // W2: ko in [0,16), nt in [0,16)
        int tt = tile - 256;
        int ko = tt >> 4, nt = tt & 15;
        int n  = nt * 16 + (lane & 15);
        int kb = ko * 32 + ((lane >> 4) << 4);
        uint32_t* dst = ws + tile * 256 + lane * 8;
#pragma unroll
        for (int i = 0; i < 8; ++i) {
            int k = kb + 2 * i;
            dst[i] = pk2bf(W2[(long)k * D_MODEL + n], W2[(long)(k + 1) * D_MODEL + n]);
        }
    }
}

// ---------------------------------------------------------------------------
// Fused: gather + LN1 -> GEMM1(bf16 WMMA) + GELU -> GEMM2 + residual -> LN2
// -> scatter.  One block (8 wave32) per 32 rows.  A operands are staged in
// LDS in WMMA fragment order: a fragment load is 2x ds_load_b128 per lane.
// ---------------------------------------------------------------------------
__global__ __launch_bounds__(256) void tree_ffn_kernel(
    const float* __restrict__ h,
    const float* __restrict__ b1,
    const float* __restrict__ b2,
    const float* __restrict__ ln_g,
    const float* __restrict__ ln_b,
    const int*  __restrict__ parent_idx,
    const int*  __restrict__ child_idx,
    const uint32_t* __restrict__ wpack,
    float* __restrict__ out,
    int E)
{
    extern __shared__ char smem[];
    float* ctx_f = (float*)(smem + CTX_F_OFF);   // [32][256] f32 (residual / ff)

    const int lane = threadIdx.x & 31;
    const int wave = threadIdx.x >> 5;
    const int row0 = blockIdx.x * TILE_M;

    // ---- Stage 1: gather parent+child, LayerNorm -> ctx (f32 + bf16 frags) --
#pragma unroll
    for (int r4 = 0; r4 < 4; ++r4) {
        int row = wave * 4 + r4;
        int ge  = row0 + row;
        int gc  = ge < E ? ge : E - 1;
        long pi = parent_idx[gc];
        long ci = child_idx[gc];
        const float4* hp = (const float4*)(h + pi * (long)D_MODEL);
        const float4* hc = (const float4*)(h + ci * (long)D_MODEL);
        float4 a0 = hp[lane * 2 + 0], a1 = hp[lane * 2 + 1];
        float4 c0 = hc[lane * 2 + 0], c1 = hc[lane * 2 + 1];
        float s[8] = { a0.x + c0.x, a0.y + c0.y, a0.z + c0.z, a0.w + c0.w,
                       a1.x + c1.x, a1.y + c1.y, a1.z + c1.z, a1.w + c1.w };
        float ls = 0.f, lq = 0.f;
#pragma unroll
        for (int j = 0; j < 8; ++j) { ls += s[j]; lq += s[j] * s[j]; }
        float mean = wsum(ls) * (1.0f / D_MODEL);
        float var  = wsum(lq) * (1.0f / D_MODEL) - mean * mean;
        float sc   = __frsqrt_rn(var + LN_EPS);
        float v[8];
#pragma unroll
        for (int j = 0; j < 8; ++j) {
            int c = lane * 8 + j;
            v[j] = (s[j] - mean) * sc * ln_g[c] + ln_b[c];
        }
        // f32 copy for the residual path
        float* cf = ctx_f + row * D_MODEL + lane * 8;
#pragma unroll
        for (int j = 0; j < 8; ++j) cf[j] = v[j];
        // bf16 copy straight into WMMA A-fragment order:
        // this thread's 8 contiguous K values = one 16 B chunk of one fragment
        int rl = row & 15, mt = row >> 4;
        int ko = lane >> 2;              // K chunk of 32
        int g  = lane & 3;               // 8-wide K group within chunk
        int lanep = rl + ((g >> 1) << 4); // consuming lane slot
        uint4 pk = make_uint4(pk2bf(v[0], v[1]), pk2bf(v[2], v[3]),
                              pk2bf(v[4], v[5]), pk2bf(v[6], v[7]));
        *(uint4*)(smem + CTX_BT_OFF + (mt * 8 + ko) * 1024
                  + lanep * 32 + ((g & 1) << 4)) = pk;
    }
    __syncthreads();

    // ---- Stage 2: GEMM1 (ctx[32x256] @ W1[256x512]) + bias + exact GELU ----
    for (int t = wave; t < 64; t += 8) {
        int mt = t >> 5;     // M tile (16 rows)
        int nt = t & 31;     // N tile (16 cols)
        v8f acc = {};
        const uint4* abase = (const uint4*)(smem + CTX_BT_OFF + mt * 8 * 1024 + lane * 32);
#pragma unroll
        for (int ko = 0; ko < 8; ++ko) {
            FragAB fa, fb;
            fa.q[0] = abase[ko * 64 + 0];          // +1024 B per ko (imm offset)
            fa.q[1] = abase[ko * 64 + 1];
            const uint4* bp = (const uint4*)(wpack + (ko * 32 + nt) * 256 + lane * 8);
            fb.q[0] = bp[0];
            fb.q[1] = bp[1];
            acc = __builtin_amdgcn_wmma_f32_16x16x32_bf16(false, fa.v, false, fb.v,
                                                          (short)0, acc, false, false);
        }
        int n    = nt * 16 + (lane & 15);
        int rl0  = (lane >> 4) << 3;               // row offset within M tile
        float bias = b1[n];
        // hid element (row=mt*16+rl0+r, col=n) -> fragment slot address
        int ko_h = n >> 5;
        int kc   = n & 31;
        int hoff = HID_BT_OFF + (mt * 16 + ko_h) * 1024 + rl0 * 32
                 + ((kc >> 4) << 9)                // +16 lane slots
                 + (((kc >> 3) & 1) << 4)          // dword group 4..7
                 + (((kc & 7) >> 1) << 2)          // dword within group
                 + ((kc & 1) << 1);                // half
#pragma unroll
        for (int r = 0; r < 8; ++r) {
            float gl = gelu_exact(acc[r] + bias);
            *(uint16_t*)(smem + hoff + r * 32) = f2bf(gl);
        }
    }
    __syncthreads();

    // ---- Stage 3: GEMM2 (hid[32x512] @ W2[512x256]) + bias + residual ----
    const uint32_t* w2pack = wpack + 256 * 256;    // after 256 W1 tiles
    for (int t = wave; t < 32; t += 8) {
        int mt = t >> 4;
        int nt = t & 15;
        v8f acc = {};
        const uint4* abase = (const uint4*)(smem + HID_BT_OFF + mt * 16 * 1024 + lane * 32);
#pragma unroll
        for (int ko = 0; ko < 16; ++ko) {
            FragAB fa, fb;
            fa.q[0] = abase[ko * 64 + 0];
            fa.q[1] = abase[ko * 64 + 1];
            const uint4* bp = (const uint4*)(w2pack + (ko * 16 + nt) * 256 + lane * 8);
            fb.q[0] = bp[0];
            fb.q[1] = bp[1];
            acc = __builtin_amdgcn_wmma_f32_16x16x32_bf16(false, fa.v, false, fb.v,
                                                          (short)0, acc, false, false);
        }
        int n     = nt * 16 + (lane & 15);
        int rbase = mt * 16 + ((lane >> 4) << 3);
        float bias = b2[n];
#pragma unroll
        for (int r = 0; r < 8; ++r) {
            float v = acc[r] + bias + ctx_f[(rbase + r) * D_MODEL + n];
            ctx_f[(rbase + r) * D_MODEL + n] = v;   // ff, in place (single RMW)
        }
    }
    __syncthreads();

    // ---- Stage 4: LayerNorm(ff) -> scatter to out[child_idx] ----
#pragma unroll
    for (int r4 = 0; r4 < 4; ++r4) {
        int row = wave * 4 + r4;
        int ge  = row0 + row;
        const float* f = ctx_f + row * D_MODEL + lane * 8;
        float s[8];
        float ls = 0.f, lq = 0.f;
#pragma unroll
        for (int j = 0; j < 8; ++j) { s[j] = f[j]; ls += s[j]; lq += s[j] * s[j]; }
        float mean = wsum(ls) * (1.0f / D_MODEL);
        float var  = wsum(lq) * (1.0f / D_MODEL) - mean * mean;
        float sc   = __frsqrt_rn(var + LN_EPS);
        if (ge < E) {
            long ci = child_idx[ge];
            float o[8];
#pragma unroll
            for (int j = 0; j < 8; ++j) {
                int c = lane * 8 + j;
                o[j] = (s[j] - mean) * sc * ln_g[c] + ln_b[c];
            }
            float4* op = (float4*)(out + ci * (long)D_MODEL + lane * 8);
            op[0] = make_float4(o[0], o[1], o[2], o[3]);
            op[1] = make_float4(o[4], o[5], o[6], o[7]);
        }
    }
}

// ---------------------------------------------------------------------------
extern "C" void kernel_launch(void* const* d_in, const int* in_sizes, int n_in,
                              void* d_out, int out_size, void* d_ws, size_t ws_size,
                              hipStream_t stream) {
    const float* h    = (const float*)d_in[0];
    const float* W1   = (const float*)d_in[1];
    const float* b1   = (const float*)d_in[2];
    const float* W2   = (const float*)d_in[3];
    const float* b2   = (const float*)d_in[4];
    const float* ln_g = (const float*)d_in[5];
    const float* ln_b = (const float*)d_in[6];
    const int*  pidx  = (const int*)d_in[7];
    const int*  cidx  = (const int*)d_in[8];
    float* out = (float*)d_out;

    int  E    = in_sizes[7];             // number of edges (= N_NODES)
    long n_h4 = (long)in_sizes[0] / 4;   // h element count / 4 (float4)

    // 1) out = h  (scatter base: non-child rows keep their value)
    copy_h<<<4096, 256, 0, stream>>>((const float4*)h, (float4*)out, n_h4);

    // 2) pack W1/W2 into bf16 WMMA-fragment tiles in workspace (512 KB)
    pack_weights<<<64, 256, 0, stream>>>(W1, W2, (uint32_t*)d_ws);

    // 3) fused gather + LN + FFN(WMMA bf16) + LN + scatter
    int blocks = (E + TILE_M - 1) / TILE_M;
    tree_ffn_kernel<<<blocks, 256, LDS_BYTES, stream>>>(h, b1, b2, ln_g, ln_b,
                                                        pidx, cidx,
                                                        (const uint32_t*)d_ws, out, E);
}
